// Pointer_decoder_40492951667375
// MI455X (gfx1250) — matrix operand order repaired and checked by reading, not compiled, not run on hardware
//
#include <hip/hip_runtime.h>
#include <hip/hip_bf16.h>
#include <math.h>

#define Bsz 256
#define Lsz 128
#define Hsz 256
#define H4  1024
#define TAUF 3.0f
#define KTC 8          // K=256 -> 8 k-tiles of 32 for every GEMM in this model

// d_out layout (floats), reference return order:
// positions (B,L) | mask_scores (B,L,L) | s0 (B,L,H) | s1 (B,L,H) | ilist (B,L,H)
#define POS_OFF 0
#define MS_OFF  (Bsz * Lsz)
#define S0_OFF  (MS_OFF + Bsz * Lsz * Lsz)
#define S1_OFF  (S0_OFF + Bsz * Lsz * Hsz)
#define IL_OFF  (S1_OFF + Bsz * Lsz * Hsz)

typedef __attribute__((ext_vector_type(16))) __bf16 v16bf;
typedef __attribute__((ext_vector_type(8)))  float  v8f;

__device__ __forceinline__ unsigned short f2bf(float f) {
  union { float f; unsigned u; } c; c.f = f;
  unsigned u = c.u;
  unsigned r = u + 0x7fffu + ((u >> 16) & 1u);   // round-to-nearest-even
  return (unsigned short)(r >> 16);
}

__device__ __forceinline__ float sigmoidf_(float x) {
  return 1.0f / (1.0f + __expf(-x));
}

// ---- WMMA fragment-major packing (CDNA5 ISA 16-bit layouts, CDNA5 ISA 7.12.2) ----
// Each 16x32 A tile / 32x16 B tile = 512 bf16 = 32 lanes x 16 contiguous halves.
__device__ __forceinline__ size_t a_pack_idx(int row, int k, int K) {
  int rt = row >> 4, m = row & 15;
  int kt = k >> 5,  kk = k & 31;
  int lane, i;
  if (kk < 8)       { lane = m;      i = kk; }
  else if (kk < 16) { lane = m + 16; i = kk - 8; }
  else if (kk < 24) { lane = m;      i = 8 + (kk - 16); }
  else              { lane = m + 16; i = 8 + (kk - 24); }
  return (((size_t)rt * (K >> 5)) + kt) * 512 + (size_t)lane * 16 + i;
}
// B tile element (kk, n): lane = n + ((kk>>3)&1 ? 16 : 0), i = (kk & 7) + (kk>=16 ? 8 : 0).
// Tile order: n-tile major, k-tile minor (streams in k for a fixed column strip).
__device__ __forceinline__ size_t b_pack_idx(int k, int n, int K) {
  int nt = n >> 4, nn = n & 15;
  int kt = k >> 5, kk = k & 31;
  int lane = nn + ((kk & 8) ? 16 : 0);
  int i = (kk & 7) + ((kk & 16) ? 8 : 0);
  return (((size_t)nt * (K >> 5)) + kt) * 512 + (size_t)lane * 16 + i;
}

// ---- pack kernels (one-time) ----
__global__ void pack_a_from_f32_kernel(const float* __restrict__ in,
                                       unsigned short* __restrict__ out,
                                       int M, int K) {
  int i = blockIdx.x * blockDim.x + threadIdx.x;
  if (i >= M * K) return;
  int row = i / K, k = i - row * K;
  out[a_pack_idx(row, k, K)] = f2bf(in[i]);
}

// Source: if transposed, in is N x K row-major and B[k][n] = in[n*K+k];
//         else in is K x N row-major and B[k][n] = in[k*N+n].
__global__ void pack_b_from_f32_kernel(const float* __restrict__ in,
                                       unsigned short* __restrict__ out,
                                       int K, int N, int transposed) {
  int i = blockIdx.x * blockDim.x + threadIdx.x;
  if (i >= K * N) return;
  int k = i / N, n = i - k * N;
  float v = transposed ? in[(size_t)n * K + k] : in[i];
  out[b_pack_idx(k, n, K)] = f2bf(v);
}

// Generic GEMM: D = A*Bm (+ A2*B2 if DUAL) (+ bias [+ bias2 if DUAL]).
// Operands pre-packed fragment-major bf16; K = KT*32 compile-time so the
// k-loop fully unrolls and every fragment load is base + immediate offset.
// One wave computes a 16x64 strip (4 WMMA tiles share one A fragment per
// k-step); 8 waves per 256-thread block.
template <int KT, bool DUAL>
__global__ void wmma_gemm_kernel(const unsigned short* __restrict__ A,
                                 const unsigned short* __restrict__ Bm,
                                 const unsigned short* __restrict__ A2,
                                 const unsigned short* __restrict__ B2,
                                 const float* __restrict__ bias,
                                 const float* __restrict__ bias2,
                                 float* __restrict__ D,
                                 int M, int N) {
  int lane = threadIdx.x & 31;
  int wave = threadIdx.x >> 5;
  int nGroups = N >> 6;                       // strips of 4 n-tiles
  int tile = blockIdx.x * 8 + wave;
  int total = (M >> 4) * nGroups;
  if (tile >= total) return;                  // wave-uniform: EXEC stays all-ones
  int tm = tile / nGroups;
  int tg = tile - tm * nGroups;
  int row0 = tm << 4, col0 = tg << 6;

  const v16bf* Afr  = (const v16bf*)A + (size_t)tm * KT * 32 + lane;
  const v16bf* Bfr  = (const v16bf*)Bm + (size_t)(tg * 4) * KT * 32 + lane;
  const v16bf* Afr2 = DUAL ? (const v16bf*)A2 + (size_t)tm * KT * 32 + lane : nullptr;
  const v16bf* Bfr2 = DUAL ? (const v16bf*)B2 + (size_t)(tg * 4) * KT * 32 + lane : nullptr;

  v8f acc0 = {0.f,0.f,0.f,0.f,0.f,0.f,0.f,0.f};
  v8f acc1 = acc0, acc2 = acc0, acc3 = acc0;

#pragma unroll
  for (int kt = 0; kt < KT; ++kt) {
    v16bf a  = Afr[kt * 32];
    v16bf b0 = Bfr[(0 * KT + kt) * 32];
    v16bf b1 = Bfr[(1 * KT + kt) * 32];
    v16bf b2 = Bfr[(2 * KT + kt) * 32];
    v16bf b3 = Bfr[(3 * KT + kt) * 32];
    acc0 = __builtin_amdgcn_wmma_f32_16x16x32_bf16(false, a, false, b0, (short)0, acc0, false, false);
    acc1 = __builtin_amdgcn_wmma_f32_16x16x32_bf16(false, a, false, b1, (short)0, acc1, false, false);
    acc2 = __builtin_amdgcn_wmma_f32_16x16x32_bf16(false, a, false, b2, (short)0, acc2, false, false);
    acc3 = __builtin_amdgcn_wmma_f32_16x16x32_bf16(false, a, false, b3, (short)0, acc3, false, false);
    if (DUAL) {
      v16bf a2 = Afr2[kt * 32];
      v16bf c0 = Bfr2[(0 * KT + kt) * 32];
      v16bf c1 = Bfr2[(1 * KT + kt) * 32];
      v16bf c2 = Bfr2[(2 * KT + kt) * 32];
      v16bf c3 = Bfr2[(3 * KT + kt) * 32];
      acc0 = __builtin_amdgcn_wmma_f32_16x16x32_bf16(false, a2, false, c0, (short)0, acc0, false, false);
      acc1 = __builtin_amdgcn_wmma_f32_16x16x32_bf16(false, a2, false, c1, (short)0, acc1, false, false);
      acc2 = __builtin_amdgcn_wmma_f32_16x16x32_bf16(false, a2, false, c2, (short)0, acc2, false, false);
      acc3 = __builtin_amdgcn_wmma_f32_16x16x32_bf16(false, a2, false, c3, (short)0, acc3, false, false);
    }
  }

  // C/D layout: column = lane&15 (+16*tile), rows i -> M = (lane<16 ? i : 8+i)
  int n  = lane & 15;
  int hi = (lane >> 4) & 1;
  int mbase = hi ? 8 : 0;
#pragma unroll
  for (int tnn = 0; tnn < 4; ++tnn) {
    int col = col0 + tnn * 16 + n;
    float bval = 0.f;
    if (bias)  bval += bias[col];
    if (DUAL && bias2) bval += bias2[col];
    v8f acc = (tnn == 0) ? acc0 : (tnn == 1) ? acc1 : (tnn == 2) ? acc2 : acc3;
    float* d = D + (size_t)(row0 + mbase) * N + col;
#pragma unroll
    for (int i = 0; i < 8; ++i) d[(size_t)i * N] = acc[i] + bval;
  }
}

// ---- state / elementwise kernels ----
__global__ void init_state_kernel(const float* __restrict__ enc,
                                  const float* __restrict__ first_state,
                                  float* __restrict__ h, float* __restrict__ c,
                                  float* __restrict__ x,
                                  unsigned short* __restrict__ hb,
                                  unsigned short* __restrict__ xb,
                                  float* __restrict__ mask) {
  int i = blockIdx.x * blockDim.x + threadIdx.x;
  if (i < Bsz * Hsz) {
    int b = i >> 8, j = i & (Hsz - 1);
    float s = 0.f;
    const float* p = enc + ((size_t)b * Lsz) * Hsz + j;
    for (int l = 0; l < Lsz; ++l) s += p[(size_t)l * Hsz];
    c[i] = s; x[i] = s;
    float hv = first_state[j];
    h[i] = hv;
    size_t pi = a_pack_idx(b, j, Hsz);
    xb[pi] = f2bf(s);
    hb[pi] = f2bf(hv);
  }
  if (i < Bsz * Lsz) mask[i] = 0.f;
}

// Record (h,c,x) into outputs at step t, then LSTM cell update in place.
// h2 also written as packed-A bf16 fragment for the q_g GEMM.
__global__ void lstm_update_kernel(const float* __restrict__ gates,
                                   float* __restrict__ h, float* __restrict__ c,
                                   const float* __restrict__ x,
                                   unsigned short* __restrict__ hb,
                                   float* __restrict__ out, int t) {
  int i = blockIdx.x * blockDim.x + threadIdx.x;
  if (i >= Bsz * Hsz) return;
  int b = i >> 8, j = i & (Hsz - 1);
  float h_old = h[i], c_old = c[i], x_old = x[i];
  size_t o = (size_t)(b * Lsz + t) * Hsz + j;
  out[S0_OFF + o] = h_old;
  out[S1_OFF + o] = c_old;
  out[IL_OFF + o] = x_old;
  const float* g = gates + (size_t)b * H4;
  float ig = sigmoidf_(g[j]);
  float fg = sigmoidf_(g[j + Hsz]);
  float gg = tanhf(g[j + 2 * Hsz]);
  float og = sigmoidf_(g[j + 3 * Hsz]);
  float c2 = fg * c_old + ig * gg;
  float h2 = og * tanhf(c2);
  c[i] = c2; h[i] = h2;
  hb[a_pack_idx(b, j, Hsz)] = f2bf(h2);
}

// scores_g + masked softmax over L, one block (128 threads) per batch row
__global__ void attn_g_kernel(const float* __restrict__ enc_g,
                              const float* __restrict__ q_g,
                              const float* __restrict__ v_g,
                              const float* __restrict__ mask,
                              float* __restrict__ attn) {
  __shared__ float red[Lsz];
  int b = blockIdx.x, l = threadIdx.x;
  const float* eg = enc_g + (size_t)(b * Lsz + l) * Hsz;
  const float* q  = q_g + (size_t)b * Hsz;
  float s = 0.f;
  for (int hh = 0; hh < Hsz; ++hh) s += v_g[hh] * tanhf(eg[hh] + q[hh]);
  s -= 1e8f * mask[b * Lsz + l];
  red[l] = s; __syncthreads();
  for (int off = Lsz / 2; off > 0; off >>= 1) {
    if (l < off) red[l] = fmaxf(red[l], red[l + off]);
    __syncthreads();
  }
  float mx = red[0]; __syncthreads();
  float e = __expf(s - mx);
  red[l] = e; __syncthreads();
  for (int off = Lsz / 2; off > 0; off >>= 1) {
    if (l < off) red[l] += red[l + off];
    __syncthreads();
  }
  attn[b * Lsz + l] = e / red[0];
}

// glimpse[b,h] = sum_l attn[b,l]*enc[b,l,h] + h2[b,h]  (per-batch B operand -> VALU)
__global__ void glimpse_kernel(const float* __restrict__ attn,
                               const float* __restrict__ enc,
                               const float* __restrict__ h,
                               float* __restrict__ gl,
                               unsigned short* __restrict__ glb) {
  int i = blockIdx.x * blockDim.x + threadIdx.x;
  if (i >= Bsz * Hsz) return;
  int b = i >> 8, j = i & (Hsz - 1);
  const float* a = attn + b * Lsz;
  const float* e = enc + (size_t)b * Lsz * Hsz + j;
  float s = 0.f;
  for (int l = 0; l < Lsz; ++l) s += a[l] * e[(size_t)l * Hsz];
  s += h[i];
  gl[i] = s;
  glb[a_pack_idx(b, j, Hsz)] = f2bf(s);
}

// final scores, mask_scores output, Gumbel argmax, mask update, pos[0] capture
__global__ void scores_argmax_kernel(const float* __restrict__ enc_p,
                                     const float* __restrict__ q,
                                     const float* __restrict__ v,
                                     float* __restrict__ mask,
                                     const float* __restrict__ u1,
                                     const float* __restrict__ u2,
                                     float* __restrict__ out,
                                     int* __restrict__ pos0, int t) {
  __shared__ float redv[Lsz];
  __shared__ int   redi[Lsz];
  int b = blockIdx.x, l = threadIdx.x;
  const float* e  = enc_p + (size_t)(b * Lsz + l) * Hsz;
  const float* qq = q + (size_t)b * Hsz;
  float s = 0.f;
  for (int hh = 0; hh < Hsz; ++hh) s += v[hh] * tanhf(e[hh] + qq[hh]);
  float masked = s - 1e7f * mask[b * Lsz + l];
  out[MS_OFF + (size_t)(b * Lsz + t) * Lsz + l] = masked;
  size_t ui = (size_t)(t * Bsz + b) * Lsz + l;
  float g1 = -__logf(-__logf(u1[ui]));
  float g2 = -__logf(-__logf(u2[ui]));
  float key = (masked + g1) / TAUF + g2;
  redv[l] = key; redi[l] = l; __syncthreads();
  for (int off = Lsz / 2; off > 0; off >>= 1) {
    if (l < off && redv[l + off] > redv[l]) { redv[l] = redv[l + off]; redi[l] = redi[l + off]; }
    __syncthreads();
  }
  if (l == 0) {
    int p = redi[0];
    out[POS_OFF + b * Lsz + t] = (float)p;
    mask[b * Lsz + p] += 1.f;
    if (b == 0) *pos0 = p;
  }
}

// inp2 = encoder_output[:, pos[0], :]  (batch-0's position for all rows)
__global__ void next_input_kernel(const float* __restrict__ enc,
                                  const int* __restrict__ pos0,
                                  float* __restrict__ x,
                                  unsigned short* __restrict__ xb) {
  int i = blockIdx.x * blockDim.x + threadIdx.x;
  if (i >= Bsz * Hsz) return;
  int b = i >> 8, j = i & (Hsz - 1);
  int p = *pos0;
  float val = enc[(size_t)(b * Lsz + p) * Hsz + j];
  x[i] = val;
  xb[a_pack_idx(b, j, Hsz)] = f2bf(val);
}

extern "C" void kernel_launch(void* const* d_in, const int* in_sizes, int n_in,
                              void* d_out, int out_size, void* d_ws, size_t ws_size,
                              hipStream_t stream) {
  const float* enc       = (const float*)d_in[0];
  const float* first_st  = (const float*)d_in[1];
  const float* W_ref_g_w = (const float*)d_in[2];
  const float* W_ref_g_b = (const float*)d_in[3];
  const float* W_q_g     = (const float*)d_in[4];
  const float* v_g       = (const float*)d_in[5];
  const float* W_ref_w   = (const float*)d_in[6];
  const float* W_ref_b   = (const float*)d_in[7];
  const float* W_q       = (const float*)d_in[8];
  const float* v_        = (const float*)d_in[9];
  const float* W_ih      = (const float*)d_in[10];
  const float* W_hh      = (const float*)d_in[11];
  const float* b_ih      = (const float*)d_in[12];
  const float* b_hh      = (const float*)d_in[13];
  const float* u1        = (const float*)d_in[14];
  const float* u2        = (const float*)d_in[15];
  float* out = (float*)d_out;

  size_t off = 0;
  auto alloc = [&](size_t bytes) -> void* {
    void* p = (char*)d_ws + off;
    off += (bytes + 255) & ~(size_t)255;
    return p;
  };
  unsigned short* encB  = (unsigned short*)alloc((size_t)Bsz * Lsz * Hsz * 2);
  unsigned short* WrgT  = (unsigned short*)alloc((size_t)Hsz * Hsz * 2);
  unsigned short* WrT   = (unsigned short*)alloc((size_t)Hsz * Hsz * 2);
  unsigned short* WqgB  = (unsigned short*)alloc((size_t)Hsz * Hsz * 2);
  unsigned short* WqB   = (unsigned short*)alloc((size_t)Hsz * Hsz * 2);
  unsigned short* WihT  = (unsigned short*)alloc((size_t)Hsz * H4 * 2);
  unsigned short* WhhT  = (unsigned short*)alloc((size_t)Hsz * H4 * 2);
  float* enc_g = (float*)alloc((size_t)Bsz * Lsz * Hsz * 4);
  float* enc_p = (float*)alloc((size_t)Bsz * Lsz * Hsz * 4);
  float* gates = (float*)alloc((size_t)Bsz * H4 * 4);
  float* hbuf  = (float*)alloc((size_t)Bsz * Hsz * 4);
  float* cbuf  = (float*)alloc((size_t)Bsz * Hsz * 4);
  float* xbuf  = (float*)alloc((size_t)Bsz * Hsz * 4);
  float* qg    = (float*)alloc((size_t)Bsz * Hsz * 4);
  float* qbuf  = (float*)alloc((size_t)Bsz * Hsz * 4);
  float* gl    = (float*)alloc((size_t)Bsz * Hsz * 4);
  unsigned short* hb  = (unsigned short*)alloc((size_t)Bsz * Hsz * 2);
  unsigned short* xb  = (unsigned short*)alloc((size_t)Bsz * Hsz * 2);
  unsigned short* glb = (unsigned short*)alloc((size_t)Bsz * Hsz * 2);
  float* attn = (float*)alloc((size_t)Bsz * Lsz * 4);
  float* mask = (float*)alloc((size_t)Bsz * Lsz * 4);
  int*   pos0 = (int*)alloc(256);

  // one-time: packing / state init / enc projections
  {
    int n = Bsz * Lsz * Hsz;
    pack_a_from_f32_kernel<<<(n + 255) / 256, 256, 0, stream>>>(enc, encB, Bsz * Lsz, Hsz);
  }
  pack_b_from_f32_kernel<<<(Hsz * Hsz + 255) / 256, 256, 0, stream>>>(W_ref_g_w, WrgT, Hsz, Hsz, 1);
  pack_b_from_f32_kernel<<<(Hsz * Hsz + 255) / 256, 256, 0, stream>>>(W_ref_w,   WrT,  Hsz, Hsz, 1);
  pack_b_from_f32_kernel<<<(Hsz * Hsz + 255) / 256, 256, 0, stream>>>(W_q_g,     WqgB, Hsz, Hsz, 0);
  pack_b_from_f32_kernel<<<(Hsz * Hsz + 255) / 256, 256, 0, stream>>>(W_q,       WqB,  Hsz, Hsz, 0);
  pack_b_from_f32_kernel<<<(Hsz * H4 + 255) / 256, 256, 0, stream>>>(W_ih, WihT, Hsz, H4, 1);
  pack_b_from_f32_kernel<<<(Hsz * H4 + 255) / 256, 256, 0, stream>>>(W_hh, WhhT, Hsz, H4, 1);
  init_state_kernel<<<(Bsz * Hsz + 255) / 256, 256, 0, stream>>>(enc, first_st, hbuf, cbuf, xbuf, hb, xb, mask);

  {
    int waves = (Bsz * Lsz / 16) * (Hsz / 64);   // 8192 strips
    wmma_gemm_kernel<KTC, false><<<waves / 8, 256, 0, stream>>>(
        encB, WrgT, nullptr, nullptr, W_ref_g_b, nullptr, enc_g, Bsz * Lsz, Hsz);
    wmma_gemm_kernel<KTC, false><<<waves / 8, 256, 0, stream>>>(
        encB, WrT, nullptr, nullptr, W_ref_b, nullptr, enc_p, Bsz * Lsz, Hsz);
  }

  int gatesWaves = (Bsz / 16) * (H4 / 64);   // 256
  int smallWaves = (Bsz / 16) * (Hsz / 64);  // 64

  for (int t = 0; t < Lsz; ++t) {
    wmma_gemm_kernel<KTC, true><<<gatesWaves / 8, 256, 0, stream>>>(
        xb, WihT, hb, WhhT, b_ih, b_hh, gates, Bsz, H4);
    lstm_update_kernel<<<(Bsz * Hsz + 255) / 256, 256, 0, stream>>>(gates, hbuf, cbuf, xbuf, hb, out, t);
    wmma_gemm_kernel<KTC, false><<<smallWaves / 8, 256, 0, stream>>>(
        hb, WqgB, nullptr, nullptr, nullptr, nullptr, qg, Bsz, Hsz);
    attn_g_kernel<<<Bsz, Lsz, 0, stream>>>(enc_g, qg, v_g, mask, attn);
    glimpse_kernel<<<(Bsz * Hsz + 255) / 256, 256, 0, stream>>>(attn, enc, hbuf, gl, glb);
    wmma_gemm_kernel<KTC, false><<<smallWaves / 8, 256, 0, stream>>>(
        glb, WqB, nullptr, nullptr, nullptr, nullptr, qbuf, Bsz, Hsz);
    scores_argmax_kernel<<<Bsz, Lsz, 0, stream>>>(enc_p, qbuf, v_, mask, u1, u2, out, pos0, t);
    next_input_kernel<<<(Bsz * Hsz + 255) / 256, 256, 0, stream>>>(enc, pos0, xbuf, xb);
  }
}
